// AttentionBlock_35330400977037
// MI455X (gfx1250) — matrix operand order repaired and checked by reading, not compiled, not run on hardware
//
#include <hip/hip_runtime.h>
#include <hip/hip_bf16.h>
#include <math.h>

typedef _Float16 v8h  __attribute__((ext_vector_type(8)));
typedef _Float16 v16h __attribute__((ext_vector_type(16)));
typedef float    v8f  __attribute__((ext_vector_type(8)));

#define HH    64
#define WWID  128
#define PP    8192      // H*W
#define CC    256
#define HID   512
#define NHEAD 8
#define HDIM  32
#define KNB   7

#define LDB   264       // 256 + 8-half pad: staggers 528B row stride across banks
#define KCHUNK 256

union AFrag { v16h v; v8h h[2]; };
union BFrag { v16h v; v8h h[2]; };

// ---------------------------------------------------------------------------
// Block-cooperative WMMA GEMM core.
// Workgroup = 256 threads = 8 waves. The block owns a 256(M) x 64(N) tile:
// wave w computes rows [m_blk + w*32, +32). The 64xK B slab (B column-major,
// i.e. Bt row-major NxK) is staged K-chunk by K-chunk into LDS once per block
// and shared by all 8 waves -> 8x less B traffic from L2.
// A-fragment VGPR layout (ISA 7.12.2, 16-bit A 16x32):
//   lanes 0-15 : row=lane,    elems 0-7 = K[0..7],  elems 8-15 = K[16..23]
//   lanes 16-31: row=lane-16, elems 0-7 = K[8..15], elems 8-15 = K[24..31]
// B-fragment: n=lane&15; same K split as A (elems 0-7 = K[8h..8h+7], 8-15 = K[16+8h..]).
// ---------------------------------------------------------------------------
template <int KD>
__device__ __forceinline__ void wmma_block_32x64(
    const _Float16* __restrict__ A, const _Float16* __restrict__ Bt,
    _Float16* Bs, int m_base, int n_base, v8f c0[4], v8f c1[4])
{
  const int lane = threadIdx.x & 31;
  const int l15  = lane & 15;
  const int lhi  = lane >> 4;
  const _Float16* arow0 = A + (size_t)(m_base + l15) * KD;
  const _Float16* arow1 = A + (size_t)(m_base + 16 + l15) * KD;

#pragma unroll
  for (int kc = 0; kc < KD; kc += KCHUNK) {
    // ---- cooperative stage of 64 x 256 B slab into LDS ----
#pragma unroll
    for (int it = 0; it < 8; ++it) {
      const int idx = it * 256 + (int)threadIdx.x;   // 0..2047
      const int row = idx >> 5;                      // 0..63
      const int col = (idx & 31) * 8;                // 0..248 halves
      *(v8h*)(Bs + row * LDB + col) =
          *(const v8h*)(Bt + (size_t)(n_base + row) * KD + kc + col);
    }
    __syncthreads();

    // ---- K loop over the staged chunk ----
#pragma unroll
    for (int k0 = 0; k0 < KCHUNK; k0 += 32) {
      AFrag a0, a1;
      a0.h[0] = *(const v8h*)(arow0 + kc + k0 + lhi * 8);
      a0.h[1] = *(const v8h*)(arow0 + kc + k0 + lhi * 8 + 16);
      a1.h[0] = *(const v8h*)(arow1 + kc + k0 + lhi * 8);
      a1.h[1] = *(const v8h*)(arow1 + kc + k0 + lhi * 8 + 16);
#pragma unroll
      for (int t = 0; t < 4; ++t) {
        const _Float16* bp = Bs + (t * 16 + l15) * LDB + k0 + lhi * 8;
        BFrag b;
        b.h[0] = *(const v8h*)(bp);       // ds_load_b128
        b.h[1] = *(const v8h*)(bp + 16);  // ds_load_b128
        c0[t] = __builtin_amdgcn_wmma_f32_16x16x32_f16(
            false, a0.v, false, b.v, (short)0, c0[t], false, false);
        c1[t] = __builtin_amdgcn_wmma_f32_16x16x32_f16(
            false, a1.v, false, b.v, (short)0, c1[t], false, false);
      }
    }
    __syncthreads();   // protect Bs before next chunk overwrite
  }
}

// D-matrix layout (f32 16x16): n = lane&15; VGPR r -> row = (lane>>4)*8 + r.

// ---- GEMM 1: QKV = Xt @ WqkvT + b  -> f32 (P x 768) ------------------------
__global__ void __launch_bounds__(256)
k_gemm_qkv(const _Float16* __restrict__ A, const _Float16* __restrict__ Bt,
           const float* __restrict__ bias, float* __restrict__ out)
{
  __shared__ _Float16 Bs[64 * LDB];
  const int tiles_n = (3 * CC) / 64;               // 12
  const int bm = blockIdx.x / tiles_n, bn = blockIdx.x % tiles_n;
  const int wv = threadIdx.x >> 5;
  const int m_base = (bm * 8 + wv) * 32, n_base = bn * 64;
  v8f c0[4] = {}, c1[4] = {};
  wmma_block_32x64<CC>(A, Bt, Bs, m_base, n_base, c0, c1);
  const int l15 = threadIdx.x & 15, lhi = (threadIdx.x & 31) >> 4;
#pragma unroll
  for (int t = 0; t < 4; ++t) {
    const int n = n_base + t * 16 + l15;
    const float bn_ = bias[n];
#pragma unroll
    for (int r = 0; r < 8; ++r) {
      const int m = m_base + lhi * 8 + r;
      out[(size_t)m * (3 * CC) + n]        = c0[t][r] + bn_;
      out[(size_t)(m + 16) * (3 * CC) + n] = c1[t][r] + bn_;
    }
  }
}

// ---- GEMM 2: x1 = attn_out @ WprojT + b + x(residual, BCHW) ----------------
__global__ void __launch_bounds__(256)
k_gemm_proj(const _Float16* __restrict__ A, const _Float16* __restrict__ Bt,
            const float* __restrict__ bias, const float* __restrict__ xres,
            float* __restrict__ x1f, _Float16* __restrict__ x1h)
{
  __shared__ _Float16 Bs[64 * LDB];
  const int tiles_n = CC / 64;                     // 4
  const int bm = blockIdx.x / tiles_n, bn = blockIdx.x % tiles_n;
  const int wv = threadIdx.x >> 5;
  const int m_base = (bm * 8 + wv) * 32, n_base = bn * 64;
  v8f c0[4] = {}, c1[4] = {};
  wmma_block_32x64<CC>(A, Bt, Bs, m_base, n_base, c0, c1);
  const int l15 = threadIdx.x & 15, lhi = (threadIdx.x & 31) >> 4;
#pragma unroll
  for (int t = 0; t < 4; ++t) {
    const int n = n_base + t * 16 + l15;
    const float bn_ = bias[n];
#pragma unroll
    for (int r = 0; r < 8; ++r) {
      const int m0 = m_base + lhi * 8 + r;
      const int m1 = m0 + 16;
      const float v0 = c0[t][r] + bn_ + xres[(size_t)n * PP + m0];
      const float v1 = c1[t][r] + bn_ + xres[(size_t)n * PP + m1];
      x1f[(size_t)m0 * CC + n] = v0;
      x1f[(size_t)m1 * CC + n] = v1;
      x1h[(size_t)m0 * CC + n] = (_Float16)v0;
      x1h[(size_t)m1 * CC + n] = (_Float16)v1;
    }
  }
}

// ---- GEMM 3: h = gelu(x1 @ Wfc1T + b) -> f16 (P x HID) ---------------------
__device__ __forceinline__ float gelu_exact(float v) {
  return 0.5f * v * (1.0f + erff(v * 0.70710678118f));
}

__global__ void __launch_bounds__(256)
k_gemm_fc1(const _Float16* __restrict__ A, const _Float16* __restrict__ Bt,
           const float* __restrict__ bias, _Float16* __restrict__ h16)
{
  __shared__ _Float16 Bs[64 * LDB];
  const int tiles_n = HID / 64;                    // 8
  const int bm = blockIdx.x / tiles_n, bn = blockIdx.x % tiles_n;
  const int wv = threadIdx.x >> 5;
  const int m_base = (bm * 8 + wv) * 32, n_base = bn * 64;
  v8f c0[4] = {}, c1[4] = {};
  wmma_block_32x64<CC>(A, Bt, Bs, m_base, n_base, c0, c1);
  const int l15 = threadIdx.x & 15, lhi = (threadIdx.x & 31) >> 4;
#pragma unroll
  for (int t = 0; t < 4; ++t) {
    const int n = n_base + t * 16 + l15;
    const float bn_ = bias[n];
#pragma unroll
    for (int r = 0; r < 8; ++r) {
      const int m0 = m_base + lhi * 8 + r;
      h16[(size_t)m0 * HID + n]        = (_Float16)gelu_exact(c0[t][r] + bn_);
      h16[(size_t)(m0 + 16) * HID + n] = (_Float16)gelu_exact(c1[t][r] + bn_);
    }
  }
}

// ---- GEMM 4: y = h @ Wfc2T + b + x1, stored BCHW ---------------------------
__global__ void __launch_bounds__(256)
k_gemm_fc2(const _Float16* __restrict__ A, const _Float16* __restrict__ Bt,
           const float* __restrict__ bias, const float* __restrict__ x1f,
           float* __restrict__ out)
{
  __shared__ _Float16 Bs[64 * LDB];
  const int tiles_n = CC / 64;                     // 4
  const int bm = blockIdx.x / tiles_n, bn = blockIdx.x % tiles_n;
  const int wv = threadIdx.x >> 5;
  const int m_base = (bm * 8 + wv) * 32, n_base = bn * 64;
  v8f c0[4] = {}, c1[4] = {};
  wmma_block_32x64<HID>(A, Bt, Bs, m_base, n_base, c0, c1);   // K=512, 2 chunks
  const int l15 = threadIdx.x & 15, lhi = (threadIdx.x & 31) >> 4;
#pragma unroll
  for (int t = 0; t < 4; ++t) {
    const int n = n_base + t * 16 + l15;
    const float bn_ = bias[n];
#pragma unroll
    for (int r = 0; r < 8; ++r) {
      const int m0 = m_base + lhi * 8 + r;
      const int m1 = m0 + 16;
      out[(size_t)n * PP + m0] = c0[t][r] + bn_ + x1f[(size_t)m0 * CC + n];
      out[(size_t)n * PP + m1] = c1[t][r] + bn_ + x1f[(size_t)m1 * CC + n];
    }
  }
}

// ---- neighborhood attention: one thread per (position, head) ---------------
__global__ void k_attn(const float* __restrict__ qkv,       // (P x 768)
                       const float* __restrict__ rpb,       // (8 x 13 x 13)
                       _Float16* __restrict__ attn_out)     // (P x 256) f16
{
  const int t = blockIdx.x * blockDim.x + threadIdx.x;
  if (t >= PP * NHEAD) return;
  const int head = t & (NHEAD - 1);
  const int p    = t >> 3;
  const int h = p >> 7;          // / WWID
  const int w = p & (WWID - 1);
  int sh = h - 3; sh = sh < 0 ? 0 : (sh > HH - KNB ? HH - KNB : sh);
  int sw = w - 3; sw = sw < 0 ? 0 : (sw > WWID - KNB ? WWID - KNB : sw);

  const float* qp = qkv + (size_t)p * (3 * CC) + head * HDIM;
  float q[HDIM];
#pragma unroll
  for (int d = 0; d < HDIM; ++d) q[d] = qp[d];

  float sc[KNB * KNB];
  float mx = -1e30f;
#pragma unroll
  for (int i = 0; i < KNB; ++i) {
    const int hh = sh + i, rh = hh - h + (KNB - 1);
#pragma unroll
    for (int j = 0; j < KNB; ++j) {
      const int ww = sw + j, rw = ww - w + (KNB - 1);
      const float* kp = qkv + (size_t)(hh * WWID + ww) * (3 * CC) + CC + head * HDIM;
      __builtin_prefetch(kp + CC, 0, 1);           // warm V row -> global_prefetch_b8
      float s = 0.f;
#pragma unroll
      for (int d = 0; d < HDIM; ++d) s = fmaf(q[d], kp[d], s);
      s = s * 0.17677669529f /* 32^-0.5 */ +
          rpb[head * 169 + rh * 13 + rw];
      sc[i * KNB + j] = s;
      mx = fmaxf(mx, s);
    }
  }
  float sum = 0.f;
#pragma unroll
  for (int e = 0; e < KNB * KNB; ++e) { sc[e] = __expf(sc[e] - mx); sum += sc[e]; }
  const float inv = 1.f / sum;

  float o[HDIM];
#pragma unroll
  for (int d = 0; d < HDIM; ++d) o[d] = 0.f;
#pragma unroll
  for (int i = 0; i < KNB; ++i) {
    const int hh = sh + i;
#pragma unroll
    for (int j = 0; j < KNB; ++j) {
      const int ww = sw + j;
      const float* vp = qkv + (size_t)(hh * WWID + ww) * (3 * CC) + 2 * CC + head * HDIM;
      const float wj = sc[i * KNB + j] * inv;
#pragma unroll
      for (int d = 0; d < HDIM; ++d) o[d] = fmaf(wj, vp[d], o[d]);
    }
  }
  _Float16* op = attn_out + (size_t)p * CC + head * HDIM;
#pragma unroll
  for (int d = 0; d < HDIM; ++d) op[d] = (_Float16)o[d];
}

// ---- layout/precision conversion helpers -----------------------------------
__global__ void k_x_to_f16t(const float* __restrict__ x, _Float16* __restrict__ xt)
{ // x: (C x P) -> xt: (P x C) f16
  const int i = blockIdx.x * blockDim.x + threadIdx.x;
  if (i >= PP * CC) return;
  const int p = i / CC, c = i % CC;
  xt[i] = (_Float16)x[(size_t)c * PP + p];
}

__global__ void k_w_to_f16t(const float* __restrict__ w, _Float16* __restrict__ wt,
                            int Kd, int Nd)
{ // w: (K x N) -> wt: (N x K) f16
  const int i = blockIdx.x * blockDim.x + threadIdx.x;
  if (i >= Kd * Nd) return;
  const int n = i / Kd, k = i % Kd;
  wt[i] = (_Float16)w[(size_t)k * Nd + n];
}

// ---------------------------------------------------------------------------
extern "C" void kernel_launch(void* const* d_in, const int* in_sizes, int n_in,
                              void* d_out, int out_size, void* d_ws, size_t ws_size,
                              hipStream_t stream)
{
  const float* x      = (const float*)d_in[0];
  const float* w_qkv  = (const float*)d_in[1];
  const float* b_qkv  = (const float*)d_in[2];
  const float* rpb    = (const float*)d_in[3];
  const float* w_proj = (const float*)d_in[4];
  const float* b_proj = (const float*)d_in[5];
  const float* w_fc1  = (const float*)d_in[6];
  const float* b_fc1  = (const float*)d_in[7];
  const float* w_fc2  = (const float*)d_in[8];
  const float* b_fc2  = (const float*)d_in[9];
  float* out = (float*)d_out;

  char* ws = (char*)d_ws;
  size_t off = 0;
  auto alloc = [&](size_t bytes) -> char* {
    char* r = ws + off;
    off = (off + bytes + 255) & ~(size_t)255;
    return r;
  };
  _Float16* xt      = (_Float16*)alloc((size_t)PP * CC * 2);
  _Float16* wqkvT   = (_Float16*)alloc((size_t)(3 * CC) * CC * 2);
  _Float16* wprojT  = (_Float16*)alloc((size_t)CC * CC * 2);
  _Float16* wfc1T   = (_Float16*)alloc((size_t)HID * CC * 2);
  _Float16* wfc2T   = (_Float16*)alloc((size_t)CC * HID * 2);
  float*    qkv     = (float*)   alloc((size_t)PP * (3 * CC) * 4);
  _Float16* attn16  = (_Float16*)alloc((size_t)PP * CC * 2);
  float*    x1f     = (float*)   alloc((size_t)PP * CC * 4);
  _Float16* x1h     = (_Float16*)alloc((size_t)PP * CC * 2);
  _Float16* h16     = (_Float16*)alloc((size_t)PP * HID * 2);
  (void)ws_size;

  const int TB = 256;
  // weight / input conversions
  k_w_to_f16t<<<(CC * 3 * CC + TB - 1) / TB, TB, 0, stream>>>(w_qkv,  wqkvT,  CC,  3 * CC);
  k_w_to_f16t<<<(CC * CC     + TB - 1) / TB, TB, 0, stream>>>(w_proj, wprojT, CC,  CC);
  k_w_to_f16t<<<(CC * HID    + TB - 1) / TB, TB, 0, stream>>>(w_fc1,  wfc1T,  CC,  HID);
  k_w_to_f16t<<<(HID * CC    + TB - 1) / TB, TB, 0, stream>>>(w_fc2,  wfc2T,  HID, CC);
  k_x_to_f16t<<<(PP * CC     + TB - 1) / TB, TB, 0, stream>>>(x, xt);

  // GEMM grids: (P/256) M-blocks x tiles_n ; 256 threads (8 waves) per block
  k_gemm_qkv<<<(PP / 256) * 12, TB, 0, stream>>>(xt, wqkvT, b_qkv, qkv);

  // neighborhood attention
  k_attn<<<(PP * NHEAD) / TB, TB, 0, stream>>>(qkv, rpb, attn16);

  k_gemm_proj<<<(PP / 256) * 4, TB, 0, stream>>>(attn16, wprojT, b_proj, x, x1f, x1h);
  k_gemm_fc1 <<<(PP / 256) * 8, TB, 0, stream>>>(x1h, wfc1T, b_fc1, h16);
  k_gemm_fc2 <<<(PP / 256) * 4, TB, 0, stream>>>(h16, wfc2T, b_fc2, x1f, out);
}
